// LSTM_model_58428735094823
// MI455X (gfx1250) — matrix operand order repaired and checked by reading, not compiled, not run on hardware
//
#include <hip/hip_runtime.h>

#define H_ 512
#define B_ 256
#define T_ 256
#define D_ 64

typedef __attribute__((ext_vector_type(16))) __bf16        v16bf;
typedef __attribute__((ext_vector_type(8)))  float         v8f;
typedef __attribute__((ext_vector_type(4)))  unsigned int  v4u;

struct U256 { v4u lo, hi; };

static __device__ __forceinline__ v16bf mkfrag(v4u lo, v4u hi) {
  U256 t{lo, hi};
  return __builtin_bit_cast(v16bf, t);
}

static __device__ __forceinline__ unsigned int f2bf(float f) {
  unsigned int u = __builtin_bit_cast(unsigned int, f);
  u = u + 0x7FFFu + ((u >> 16) & 1u);   // round-to-nearest-even
  return u >> 16;
}

static __device__ __forceinline__ float sigmoidf_(float x) {
  return 1.0f / (1.0f + __expf(-x));
}
static __device__ __forceinline__ float tanhf_(float x) {
  x = fminf(fmaxf(x, -15.0f), 15.0f);
  float e = __expf(2.0f * x);
  return (e - 1.0f) / (e + 1.0f);
}

// ---------------------------------------------------------------------------
// GEMM partial: accumulate A[16 x 32*ktiles] x Wpacked into 4 gate fragments.
// A: row-major bf16, row stride `astride` elements.
//   16-bit A layout: lanes 0-15 row M=lane, v0..7 = K 0..15;
//   lanes 16-31 same rows, K 16..31 (matches packed B).
// W: packed B tiles, tile (gnt, kt) = 256 dwords at ((gnt*KT + kt)*256),
//   lane-major (lane*8 + v), dword v = {K=2v (lo16), K=2v+1 (hi16)} with
//   K-base = (lane>=16 ? 16 : 0), column N = lane%16.
// ---------------------------------------------------------------------------
static __device__ __forceinline__ void gemm_part(
    const unsigned short* __restrict__ arow, int ktiles,
    const unsigned int* __restrict__ W, int nt, int lane, int hi,
    v8f& g0, v8f& g1, v8f& g2, v8f& g3)
{
  const size_t gs = (size_t)32 * ktiles * 256;   // dword stride between gates
  for (int kt = 0; kt < ktiles; ++kt) {
    const unsigned short* p = arow + kt * 32 + hi * 16;
    v16bf A = mkfrag(*(const v4u*)p, *(const v4u*)(p + 8));
    const unsigned int* wb = W + ((size_t)nt * ktiles + kt) * 256 + lane * 8;
    v16bf Bq;
    Bq = mkfrag(((const v4u*)wb)[0], ((const v4u*)wb)[1]);
    g0 = __builtin_amdgcn_wmma_f32_16x16x32_bf16(false, A, false, Bq, (short)0, g0, false, false);
    wb += gs;
    Bq = mkfrag(((const v4u*)wb)[0], ((const v4u*)wb)[1]);
    g1 = __builtin_amdgcn_wmma_f32_16x16x32_bf16(false, A, false, Bq, (short)0, g1, false, false);
    wb += gs;
    Bq = mkfrag(((const v4u*)wb)[0], ((const v4u*)wb)[1]);
    g2 = __builtin_amdgcn_wmma_f32_16x16x32_bf16(false, A, false, Bq, (short)0, g2, false, false);
    wb += gs;
    Bq = mkfrag(((const v4u*)wb)[0], ((const v4u*)wb)[1]);
    g3 = __builtin_amdgcn_wmma_f32_16x16x32_bf16(false, A, false, Bq, (short)0, g3, false, false);
  }
}

// ---------------------------------------------------------------------------
// One LSTM time step for one layer, fused dual-GEMM + gate update.
//   gates = A0 @ W0^T + A1 @ W1^T + bias ; i,f,g,o -> c,h update.
// grid = (16 batch tiles, 4 n-blocks), block = 256 (8 waves).
// Wave w owns n-tile nt = by*8+w (columns nt*16..nt*16+15 of each gate).
// ---------------------------------------------------------------------------
__global__ __launch_bounds__(256) void lstm_step(
    const unsigned short* __restrict__ A0, int a0_stride, int k0_tiles,
    const unsigned int* __restrict__ W0,
    const unsigned short* __restrict__ A1,
    const unsigned int* __restrict__ W1,
    const float* __restrict__ bias, float* __restrict__ c_state,
    unsigned short* __restrict__ h_out, float* __restrict__ h_sum)
{
  const int lane = threadIdx.x & 31;
  const int wave = threadIdx.x >> 5;
  const int nt   = blockIdx.y * 8 + wave;   // 0..31  (column tile in H)
  const int mt   = blockIdx.x;              // 0..15  (batch tile)
  const int col  = lane & 15;
  const int hi   = lane >> 4;

  v8f gi = {0.f,0.f,0.f,0.f,0.f,0.f,0.f,0.f};
  v8f gf = {0.f,0.f,0.f,0.f,0.f,0.f,0.f,0.f};
  v8f gg = {0.f,0.f,0.f,0.f,0.f,0.f,0.f,0.f};
  v8f go = {0.f,0.f,0.f,0.f,0.f,0.f,0.f,0.f};

  const int row = mt * 16 + col;            // batch row for A fragments
  gemm_part(A0 + (size_t)row * a0_stride, k0_tiles, W0, nt, lane, hi, gi, gf, gg, go);
  gemm_part(A1 + (size_t)row * H_,        H_ / 32,  W1, nt, lane, hi, gi, gf, gg, go);

  // C/D layout: lanes 0-15 VGPR r -> M=r ; lanes 16-31 -> M=8+r ; N = lane%16.
  const int n = nt * 16 + col;              // 0..511
  const float bi = bias[n];
  const float bf2 = bias[H_ + n];
  const float bg = bias[2 * H_ + n];
  const float bo = bias[3 * H_ + n];
#pragma unroll
  for (int r = 0; r < 8; ++r) {
    const int m = mt * 16 + hi * 8 + r;
    const size_t idx = (size_t)m * H_ + n;
    float iv = sigmoidf_(gi[r] + bi);
    float fv = sigmoidf_(gf[r] + bf2);
    float gv = tanhf_(gg[r] + bg);
    float ov = sigmoidf_(go[r] + bo);
    float cv = fv * c_state[idx] + iv * gv;
    c_state[idx] = cv;
    float hv = ov * tanhf_(cv);
    h_out[idx] = (unsigned short)f2bf(hv);
    if (h_sum) h_sum[idx] += hv;
  }
}

// Pack W[4H, K] f32 -> bf16 WMMA B-operand tiles. grid=(128, K/32), block=32.
__global__ void pack_weights(const float* __restrict__ W, int K,
                             unsigned int* __restrict__ out)
{
  const int nt   = blockIdx.x;          // global 16-col tile over 4H (0..127)
  const int kt   = blockIdx.y;
  const int KT   = gridDim.y;
  const int lane = threadIdx.x;
  const int n    = nt * 16 + (lane & 15);
  const int kb   = kt * 32 + (lane >> 4) * 16;
  const float* wr = W + (size_t)n * K + kb;
  unsigned int d[8];
#pragma unroll
  for (int v = 0; v < 8; ++v)
    d[v] = f2bf(wr[2 * v]) | (f2bf(wr[2 * v + 1]) << 16);
  unsigned int* o = out + ((size_t)nt * KT + kt) * 256 + lane * 8;
  v4u* o4 = (v4u*)o;
  o4[0] = v4u{d[0], d[1], d[2], d[3]};
  o4[1] = v4u{d[4], d[5], d[6], d[7]};
}

__global__ void cvt_bf(const float* __restrict__ in,
                       unsigned short* __restrict__ out, int n)
{
  int i = (blockIdx.x * blockDim.x + threadIdx.x) * 4;
  if (i + 3 < n) {
#pragma unroll
    for (int k = 0; k < 4; ++k) out[i + k] = (unsigned short)f2bf(in[i + k]);
  }
}

__global__ void zero32(unsigned int* __restrict__ p, int n)
{
  int i = blockIdx.x * blockDim.x + threadIdx.x;
  if (i < n) p[i] = 0u;
}

// heads: lh = sum/T ; out = relu(lh @ Wsh^T + bsh)*1.5 ; dir/mag dots.
__global__ __launch_bounds__(256) void head_kernel(
    const float* __restrict__ hsum,
    const float* __restrict__ Wsh,  const float* __restrict__ bsh,
    const float* __restrict__ Wdir, const float* __restrict__ bdir,
    const float* __restrict__ Wmag, const float* __restrict__ bmag,
    float* __restrict__ out)
{
  const int b = blockIdx.x, j = threadIdx.x;
  const float* lh = hsum + (size_t)b * H_;
  const float* wr = Wsh + (size_t)j * H_;
  float acc = 0.0f;
  for (int k = 0; k < H_; ++k) acc += lh[k] * wr[k];
  float o = acc * (1.0f / (float)T_) + bsh[j];
  o = fmaxf(o, 0.0f) * 1.5f;
  __shared__ float sd[256], sm[256];
  sd[j] = o * Wdir[j];
  sm[j] = o * Wmag[j];
  __syncthreads();
  for (int s = 128; s > 0; s >>= 1) {
    if (j < s) { sd[j] += sd[j + s]; sm[j] += sm[j + s]; }
    __syncthreads();
  }
  if (j == 0) {
    out[b]       = sd[0] + bdir[0];
    out[B_ + b]  = sm[0] + bmag[0];
  }
}

extern "C" void kernel_launch(void* const* d_in, const int* in_sizes, int n_in,
                              void* d_out, int out_size, void* d_ws, size_t ws_size,
                              hipStream_t stream)
{
  const float* x      = (const float*)d_in[0];
  const float* W_ih_0 = (const float*)d_in[1];
  const float* W_hh_0 = (const float*)d_in[2];
  const float* b_0    = (const float*)d_in[3];
  const float* W_ih_1 = (const float*)d_in[4];
  const float* W_hh_1 = (const float*)d_in[5];
  const float* b_1    = (const float*)d_in[6];
  const float* W_sh   = (const float*)d_in[7];
  const float* b_sh   = (const float*)d_in[8];
  const float* W_dir  = (const float*)d_in[9];
  const float* b_dir  = (const float*)d_in[10];
  const float* W_mag  = (const float*)d_in[11];
  const float* b_mag  = (const float*)d_in[12];
  float* out = (float*)d_out;

  char* ws = (char*)d_ws;
  unsigned short* xbf    = (unsigned short*)(ws + 0);          // 8,388,608 B
  unsigned int*   pw_ih0 = (unsigned int*)(ws + 8388608);      //   262,144 B
  unsigned int*   pw_hh0 = (unsigned int*)(ws + 8650752);      // 2,097,152 B
  unsigned int*   pw_ih1 = (unsigned int*)(ws + 10747904);     // 2,097,152 B
  unsigned int*   pw_hh1 = (unsigned int*)(ws + 12845056);     // 2,097,152 B
  float*          c0     = (float*)(ws + 14942208);            //   524,288 B
  float*          c1     = (float*)(ws + 15466496);            //   524,288 B
  float*          hsum   = (float*)(ws + 15990784);            //   524,288 B
  unsigned short* h0a    = (unsigned short*)(ws + 16515072);   //   262,144 B
  unsigned short* h0b    = (unsigned short*)(ws + 16777216);
  unsigned short* h1a    = (unsigned short*)(ws + 17039360);
  unsigned short* h1b    = (unsigned short*)(ws + 17301504);   // end 17,563,648

  // zero c0,c1,hsum,h buffers (2,621,440 B = 655,360 dwords)
  zero32<<<2560, 256, 0, stream>>>((unsigned int*)(ws + 14942208), 655360);
  cvt_bf<<<4096, 256, 0, stream>>>(x, xbf, B_ * T_ * D_);
  pack_weights<<<dim3(128, 2),  32, 0, stream>>>(W_ih_0,  64, pw_ih0);
  pack_weights<<<dim3(128, 16), 32, 0, stream>>>(W_hh_0, 512, pw_hh0);
  pack_weights<<<dim3(128, 16), 32, 0, stream>>>(W_ih_1, 512, pw_ih1);
  pack_weights<<<dim3(128, 16), 32, 0, stream>>>(W_hh_1, 512, pw_hh1);

  dim3 grd(16, 4), blk(256);
  for (int t = 0; t < T_; ++t) {
    const unsigned short* h0p = (t & 1) ? h0b : h0a;
    unsigned short*       h0n = (t & 1) ? h0a : h0b;
    const unsigned short* h1p = (t & 1) ? h1b : h1a;
    unsigned short*       h1n = (t & 1) ? h1a : h1b;
    // layer 0: gates = x_t @ W_ih0^T + h0_{t-1} @ W_hh0^T + b0
    lstm_step<<<grd, blk, 0, stream>>>(xbf + (size_t)t * D_, T_ * D_, 2, pw_ih0,
                                       h0p, pw_hh0, b_0, c0, h0n, (float*)nullptr);
    // layer 1: gates = h0_t @ W_ih1^T + h1_{t-1} @ W_hh1^T + b1 ; accumulate mean
    lstm_step<<<grd, blk, 0, stream>>>(h0n, H_, 16, pw_ih1,
                                       h1p, pw_hh1, b_1, c1, h1n, hsum);
  }
  head_kernel<<<256, 256, 0, stream>>>(hsum, W_sh, b_sh, W_dir, b_dir,
                                       W_mag, b_mag, out);
  (void)in_sizes; (void)n_in; (void)out_size; (void)ws_size;
}
